// GCN_82609400971778
// MI455X (gfx1250) — matrix operand order repaired and checked by reading, not compiled, not run on hardware
//
#include <hip/hip_runtime.h>
#include <stdint.h>

// ---------- CDNA5 WMMA types ----------
typedef __attribute__((ext_vector_type(16))) __bf16 v16bf;
typedef __attribute__((ext_vector_type(8)))  __bf16 v8bf;
typedef __attribute__((ext_vector_type(8)))  float  v8f;

__device__ __forceinline__ unsigned short f32_to_bf16(float f) {
    union { float f; uint32_t u; } c; c.f = f;
    uint32_t u = c.u;
    uint32_t r = u + 0x7FFFu + ((u >> 16) & 1u);   // round-to-nearest-even
    return (unsigned short)(r >> 16);
}
__device__ __forceinline__ float bf16_to_f32(unsigned short h) {
    union { uint32_t u; float f; } c; c.u = ((uint32_t)h) << 16;
    return c.f;
}

// ---------- setup kernels ----------
__global__ __launch_bounds__(256) void k_zero_u32(unsigned int* p, int n) {
    int i = blockIdx.x * 256 + threadIdx.x;
    if (i < n) p[i] = 0u;
}

__global__ __launch_bounds__(256) void k_deg(const long long* __restrict__ dst,
                                             unsigned int* __restrict__ deg, int E) {
    int e = blockIdx.x * 256 + threadIdx.x;
    if (e < E) atomicAdd(&deg[(int)dst[e]], 1u);
}

__global__ __launch_bounds__(256) void k_dinv(const unsigned int* __restrict__ deg,
                                              float* __restrict__ dinv, int N) {
    int i = blockIdx.x * 256 + threadIdx.x;
    if (i < N) dinv[i] = rsqrtf((float)deg[i] + 1.0f);   // +1 self loop, >= 1
}

__global__ __launch_bounds__(256) void k_prep(const long long* __restrict__ ei,
                                              const float* __restrict__ dinv,
                                              int* __restrict__ s32, int* __restrict__ d32,
                                              float* __restrict__ normE, int E) {
    int e = blockIdx.x * 256 + threadIdx.x;
    if (e < E) {
        int s = (int)ei[e];
        int d = (int)ei[(size_t)E + e];
        s32[e] = s; d32[e] = d;
        normE[e] = dinv[s] * dinv[d];
    }
}

__global__ __launch_bounds__(256) void k_f2bf(const float* __restrict__ in,
                                              unsigned short* __restrict__ out, int n) {
    int i = blockIdx.x * 256 + threadIdx.x;
    if (i < n) out[i] = f32_to_bf16(in[i]);
}

// Pre-swizzle W (fp32, [din,dout]) into bf16 B-fragment layout, tile-major:
// Wsw[nt][(ks*32 + lane)*16 + j] = W[ks*32 + (lane>>4)*16 + j][nt*16 + (lane&15)]
// so a lane's 16-half B fragment is one contiguous 32-byte run.
__global__ __launch_bounds__(256) void k_swz(const float* __restrict__ W,
                                             unsigned short* __restrict__ Wsw,
                                             int din, int dout) {
    int idx = blockIdx.x * 256 + threadIdx.x;
    int total = din * dout;
    if (idx < total) {
        int tilesz = din * 16;
        int nt = idx / tilesz;
        int r  = idx - nt * tilesz;
        int j  = r & 15;
        int lr = r >> 4;
        int ks = lr >> 5;
        int lane = lr & 31;
        int g = lane >> 4, n = lane & 15;
        int k = ks * 32 + g * 16 + j;
        Wsw[idx] = f32_to_bf16(W[(size_t)k * dout + nt * 16 + n]);
    }
}

// ---------- WMMA GEMM with fused self-loop/bias epilogue ----------
// Hg[N,dout] = Hbf[N,din] * W ; Agg[N,dout] = bias + dinv^2 * Hg  (self-loop init)
// One wave -> one 16-row strip; 8 waves/block. B tiles staged in LDS via
// CDNA5 async global->LDS block copy (pre-swizzled, contiguous b128 per lane).
template<int DIN>
__global__ __launch_bounds__(256)
void k_gemm(const unsigned short* __restrict__ Abf,
            const unsigned short* __restrict__ Wsw,
            float* __restrict__ Hg, float* __restrict__ Agg,
            const float* __restrict__ bias, const float* __restrict__ dinv,
            int nrows, int dout) {
    constexpr int KS = DIN / 32;
    __shared__ __align__(32) unsigned short ldsb[KS * 32 * 16];

    const int lane  = threadIdx.x & 31;
    const int wave  = threadIdx.x >> 5;
    const int strip = blockIdx.x * 8 + wave;
    const int m = lane & 15;
    const int g = lane >> 4;
    const bool valid = strip < (nrows >> 4);

    // Preload A fragments for this 16-row strip (register-resident for all N-tiles).
    // A 16x32 bf16 layout: elem j -> K = ks*32 + g*16 + j.
    v16bf afrag[KS] = {};
    if (valid) {
        const unsigned short* arow = Abf + (size_t)(strip * 16 + m) * DIN;
#pragma unroll
        for (int ks = 0; ks < KS; ++ks) {
            union { v16bf v; struct { v8bf lo; v8bf hi; } h; } t;
            const int base = ks * 32 + g * 16;
            t.h.lo = *(const v8bf*)(arow + base);
            t.h.hi = *(const v8bf*)(arow + base + 8);
            afrag[ks] = t.v;
        }
    }

    const int ntiles = dout >> 4;
    for (int nt = 0; nt < ntiles; ++nt) {
        // Async block copy of the pre-swizzled 16-column W tile into LDS:
        // DIN*16 halfs = DIN*2 chunks of 16B; each active lane moves 16B
        // memory->LDS without touching VGPRs (tracked by ASYNCcnt).
        const unsigned short* wt = Wsw + (size_t)nt * (DIN * 16);
        for (int c = threadIdx.x; c < DIN * 2; c += 256) {
            unsigned ldsoff = (unsigned)(uintptr_t)(&ldsb[c * 8]);
            const unsigned short* gp = wt + c * 8;
            asm volatile("global_load_async_to_lds_b128 %0, %1, off"
                         :: "v"(ldsoff), "v"(gp) : "memory");
        }
        asm volatile("s_wait_asynccnt 0x0" ::: "memory");
        __syncthreads();

        if (valid) {
            v8f acc = {};
#pragma unroll
            for (int ks = 0; ks < KS; ++ks) {
                v16bf bfrag = *(const v16bf*)(&ldsb[(ks * 32 + lane) << 4]);
                acc = __builtin_amdgcn_wmma_f32_16x16x32_bf16(
                    false, afrag[ks], false, bfrag, (short)0, acc, false, false);
            }
            // D layout: col = nt*16 + m ; row = strip*16 + g*8 + v
            const int col  = nt * 16 + m;
            const int row0 = strip * 16 + g * 8;
            const float bcol = bias[col];
            float* hp = Hg  + (size_t)row0 * dout + col;
            float* ap = Agg + (size_t)row0 * dout + col;
#pragma unroll
            for (int v = 0; v < 8; ++v) {
                float dv = dinv[row0 + v];
                hp[(size_t)v * dout] = acc[v];
                ap[(size_t)v * dout] = bcol + dv * dv * acc[v];
            }
        }
        __syncthreads();
    }
}

// Last layer (dout == 1): dot product per node, fp32 weights, fused epilogue.
__global__ __launch_bounds__(256) void k_gemv(const unsigned short* __restrict__ Hbf,
                                              const float* __restrict__ W,
                                              const float* __restrict__ bias,
                                              const float* __restrict__ dinv,
                                              float* __restrict__ Hg,
                                              float* __restrict__ Agg,
                                              int N, int din) {
    int n = blockIdx.x * 256 + threadIdx.x;
    if (n < N) {
        const unsigned short* row = Hbf + (size_t)n * din;
        float s = 0.0f;
        for (int k = 0; k < din; ++k) s += bf16_to_f32(row[k]) * W[k];
        float dv = dinv[n];
        Hg[n]  = s;
        Agg[n] = bias[0] + dv * dv * s;
    }
}

// ---------- aggregation ----------
// Agg[dst,f..f+3] += normE[e] * Hg[src,f..f+3] — vectorized gather, f32 atomics.
__global__ __launch_bounds__(256) void k_scatter4(const float* __restrict__ Hg,
                                                  const int* __restrict__ s32,
                                                  const int* __restrict__ d32,
                                                  const float* __restrict__ normE,
                                                  float* __restrict__ Agg,
                                                  int total4, int dout4, int dout) {
    int i = blockIdx.x * 256 + threadIdx.x;
    if (i < total4) {
        int e = i / dout4;
        int f = (i - e * dout4) << 2;
        float nv = normE[e];
        const float4 h = *(const float4*)(Hg + (size_t)s32[e] * dout + f);
        float* a = Agg + (size_t)d32[e] * dout + f;
        atomicAdd(a + 0, nv * h.x);
        atomicAdd(a + 1, nv * h.y);
        atomicAdd(a + 2, nv * h.z);
        atomicAdd(a + 3, nv * h.w);
    }
}

__global__ __launch_bounds__(256) void k_scatter1(const float* __restrict__ Hg,
                                                  const int* __restrict__ s32,
                                                  const int* __restrict__ d32,
                                                  const float* __restrict__ normE,
                                                  float* __restrict__ Agg, int E) {
    int e = blockIdx.x * 256 + threadIdx.x;
    if (e < E) atomicAdd(&Agg[d32[e]], normE[e] * Hg[s32[e]]);
}

// ReLU (optional) + convert to bf16 for next layer's A matrix.
__global__ __launch_bounds__(256) void k_finalize(const float* __restrict__ Agg,
                                                  unsigned short* __restrict__ Hbf,
                                                  int n, int relu) {
    int i = blockIdx.x * 256 + threadIdx.x;
    if (i < n) {
        float v = Agg[i];
        if (relu) v = fmaxf(v, 0.0f);
        Hbf[i] = f32_to_bf16(v);
    }
}

__global__ __launch_bounds__(256) void k_out(const float* __restrict__ Agg,
                                             float* __restrict__ out, int N) {
    int i = blockIdx.x * 256 + threadIdx.x;
    if (i < N) out[i] = Agg[i];
}

// ---------- host ----------
static inline dim3 grid1(long long n) { return dim3((unsigned)((n + 255) / 256)); }

extern "C" void kernel_launch(void* const* d_in, const int* in_sizes, int n_in,
                              void* d_out, int out_size, void* d_ws, size_t ws_size,
                              hipStream_t stream) {
    (void)n_in; (void)out_size; (void)ws_size;
    const float*     x  = (const float*)d_in[0];
    const long long* ei = (const long long*)d_in[1];   // int64 edge_index [2,E]
    const int F_IN = 128;
    const int N = in_sizes[0] / F_IN;
    const int E = in_sizes[1] / 2;

    static const int dims[9][2] = {{128,128},{128,192},{192,256},{256,256},{256,256},
                                   {256,256},{256,192},{192,128},{128,1}};
    static const int relu[9] = {1,1,1,1,0,1,1,1,0};

    // carve workspace
    char* ws = (char*)d_ws;
    size_t off = 0;
    auto carve = [&](size_t bytes) -> void* {
        void* p = ws + off;
        off += (bytes + 255) & ~(size_t)255;
        return p;
    };
    unsigned short* Hbf  = (unsigned short*)carve((size_t)N * 256 * 2);
    float*          Hg   = (float*)carve((size_t)N * 256 * 4);
    float*          Agg  = (float*)carve((size_t)N * 256 * 4);
    unsigned short* Wsw  = (unsigned short*)carve((size_t)256 * 256 * 2);
    unsigned int*   deg  = (unsigned int*)carve((size_t)N * 4);
    float*          dinv = (float*)carve((size_t)N * 4);
    int*            s32  = (int*)carve((size_t)E * 4);
    int*            d32  = (int*)carve((size_t)E * 4);
    float*          nrmE = (float*)carve((size_t)E * 4);

    // normalization (recomputed every call; deterministic)
    k_zero_u32<<<grid1(N), 256, 0, stream>>>(deg, N);
    k_deg<<<grid1(E), 256, 0, stream>>>(ei + E, deg, E);
    k_dinv<<<grid1(N), 256, 0, stream>>>(deg, dinv, N);
    k_prep<<<grid1(E), 256, 0, stream>>>(ei, dinv, s32, d32, nrmE, E);

    // initial activations -> bf16
    k_f2bf<<<grid1((long long)N * F_IN), 256, 0, stream>>>(x, Hbf, N * F_IN);

    const int nstrips = N >> 4;
    const dim3 ggrid((nstrips + 7) / 8);

    for (int i = 0; i < 9; ++i) {
        const int din = dims[i][0], dout = dims[i][1];
        const float* W = (const float*)d_in[2 + 2 * i];
        const float* b = (const float*)d_in[3 + 2 * i];

        if (dout > 1) {
            k_swz<<<grid1((long long)din * dout), 256, 0, stream>>>(W, Wsw, din, dout);
            if (din == 128)
                k_gemm<128><<<ggrid, 256, 0, stream>>>(Hbf, Wsw, Hg, Agg, b, dinv, N, dout);
            else if (din == 192)
                k_gemm<192><<<ggrid, 256, 0, stream>>>(Hbf, Wsw, Hg, Agg, b, dinv, N, dout);
            else
                k_gemm<256><<<ggrid, 256, 0, stream>>>(Hbf, Wsw, Hg, Agg, b, dinv, N, dout);
        } else {
            k_gemv<<<grid1(N), 256, 0, stream>>>(Hbf, W, b, dinv, Hg, Agg, N, din);
        }

        // scatter neighbors into Agg (already holds bias + self-loop term)
        if ((dout & 3) == 0) {
            const int dout4 = dout >> 2;
            const long long t4 = (long long)E * dout4;
            k_scatter4<<<grid1(t4), 256, 0, stream>>>(Hg, s32, d32, nrmE, Agg,
                                                      (int)t4, dout4, dout);
        } else {
            k_scatter1<<<grid1(E), 256, 0, stream>>>(Hg, s32, d32, nrmE, Agg, E);
        }

        const long long nTot = (long long)N * dout;
        if (i < 8)
            k_finalize<<<grid1(nTot), 256, 0, stream>>>(Agg, Hbf, (int)nTot, relu[i]);
        else
            k_out<<<grid1(N), 256, 0, stream>>>(Agg, (float*)d_out, N);
    }
}
